// LinearAttentionNeuralOperator_29927332119147
// MI455X (gfx1250) — matrix operand order repaired and checked by reading, not compiled, not run on hardware
//
#include <hip/hip_runtime.h>
#include <hip/hip_bf16.h>

// LinearAttentionNeuralOperator forward for MI455X (gfx1250, wave32, WMMA).
// Big GEMMs: v_wmma_f32_16x16x32_bf16 with double-buffered LDS tiles fed by
// gfx1250 async global->LDS loads (b128/b64, s_wait_asynccnt paced). Residual
// fx in fp32, activations bf16 (HBM-leaning: ~1.4e11 FLOPs vs ~1.5-2 GB).

typedef __bf16 bf16_t;
typedef __attribute__((ext_vector_type(16))) __bf16 v16bf;
typedef __attribute__((ext_vector_type(8)))  float  v8f;
typedef int v2i __attribute__((vector_size(8)));     // b64 payload type
typedef int v4i __attribute__((vector_size(16)));    // b128 payload type

#define DEVI __device__ __forceinline__
#define AS1 __attribute__((address_space(1)))
#define AS3 __attribute__((address_space(3)))

#if defined(__has_builtin)
#  if __has_builtin(__builtin_amdgcn_global_load_async_to_lds_b64) && \
      __has_builtin(__builtin_amdgcn_global_load_async_to_lds_b128)
#    define HAS_ASYNC_LDS 1
#  endif
#endif
#ifndef HAS_ASYNC_LDS
#  define HAS_ASYNC_LDS 0
#endif

DEVI float gelu_f(float x) { return 0.5f * x * (1.0f + erff(x * 0.70710678118654752f)); }
DEVI float clampt(float t) { return fminf(fmaxf(t, 0.1f), 2.0f); }

// ---------------------------------------------------------------------------
// Generic tiled GEMM: C[M x Nout] = A[M x K] @ B   (bf16 in, f32 acc)
//  BT=1: B pre-transposed [Nout][K] (ldb = K)  -> contiguous async staging
//  BT=0: B row-major [K][Nout] (ldb = Nout)    -> transpose-scatter staging
// grid.x = M/BM, grid.y = Nout/BN, grid.z = heads * zDiv (zDiv = K-splits).
// Epilogues: 0 bias->bf16, 1 gelu(bias)->bf16, 2 bias+bias2->f32,
//            3 f32 out += acc+bias (residual), 4 plain bf16, 5 atomicAdd f32.
// ---------------------------------------------------------------------------
template<int BM, int BN, int EPI, int BT>
__launch_bounds__(256)
__global__ void gemm_k(const bf16_t* __restrict__ A, long aZStep, int lda,
                       const bf16_t* __restrict__ B, long bZStep, int ldb,
                       int K, int zDiv,
                       float* __restrict__ outF, bf16_t* __restrict__ outB,
                       long outZStep, int ldout,
                       const float* __restrict__ bias,
                       const float* __restrict__ bias2)
{
    constexpr int CT = BN / 16;
    constexpr int ACALLS = (BM * 32) / 2048;          // b128: 8-elem chunks  (=2)
    constexpr int BCALLS = 1;                         // b128 (BN=64) / b64 (BN=32)
    // per-wave async instructions per tile (uniform across all waves)
    constexpr int PERWAVE = BT ? (ACALLS + BCALLS) : ACALLS;

    __shared__ __align__(16) bf16_t As[2][BM][40];    // 32 K + pad
    __shared__ __align__(16) bf16_t Bs[2][BN][40];    // Bs[n][k]

    const int z     = blockIdx.z;
    const int head  = z / zDiv;
    const int split = z - head * zDiv;
    const long aColOff = (long)split * K;
    const long bColOff = (long)split * K;             // BT=1: col offset in K
    const bf16_t* aB = A + (long)head * aZStep;
    const bf16_t* bB = B + (long)head * bZStep
                         + (BT ? 0 : (long)split * (long)K * ldb);

    const int mBase = blockIdx.x * BM;
    const int nBase = blockIdx.y * BN;
    const int tid  = threadIdx.x;
    const int lane = tid & 31;
    const int wid  = tid >> 5;
    const int rb   = wid * 16;                        // 8 waves -> 8 row tiles

    v8f acc[CT] = {};

    auto stage = [&](int buf, int k0) {
        // ---- A tile (BM x 32), b128 chunks: 8 bf16 per lane ----
        #pragma unroll
        for (int it = 0; it < ACALLS; ++it) {
            int q = tid + it * 256;
            int r = q >> 2, c = (q & 3) << 3;
            const bf16_t* src = aB + (long)(mBase + r) * lda + aColOff + k0 + c;
#if HAS_ASYNC_LDS
            __builtin_amdgcn_global_load_async_to_lds_b128(
                (AS1 v4i*)src, (AS3 v4i*)&As[buf][r][c], 0, 0);
#else
            *reinterpret_cast<uint4*>(&As[buf][r][c]) =
                *reinterpret_cast<const uint4*>(src);
#endif
        }
        // ---- B tile ----
        if constexpr (BT) {                           // pre-transposed [Nout][K]
            if constexpr (BN == 64) {                 // 2048 elems: one b128 call
                int r = tid >> 2, c = (tid & 3) << 3;
                const bf16_t* src = bB + (long)(nBase + r) * ldb + bColOff + k0 + c;
#if HAS_ASYNC_LDS
                __builtin_amdgcn_global_load_async_to_lds_b128(
                    (AS1 v4i*)src, (AS3 v4i*)&Bs[buf][r][c], 0, 0);
#else
                *reinterpret_cast<uint4*>(&Bs[buf][r][c]) =
                    *reinterpret_cast<const uint4*>(src);
#endif
            } else {                                  // BN==32: one b64 call
                int r = tid >> 3, c = (tid & 7) << 2;
                const bf16_t* src = bB + (long)(nBase + r) * ldb + bColOff + k0 + c;
#if HAS_ASYNC_LDS
                __builtin_amdgcn_global_load_async_to_lds_b64(
                    (AS1 v2i*)src, (AS3 v2i*)&Bs[buf][r][c], 0, 0);
#else
                *reinterpret_cast<uint2*>(&Bs[buf][r][c]) =
                    *reinterpret_cast<const uint2*>(src);
#endif
            }
        } else {                                      // row-major [K][Nout]
            #pragma unroll
            for (int it = 0; it < (BN * 32) / 1024; ++it) {
                int l = tid * 4 + it * 1024;
                int kr = l / BN, c = l % BN;
                const bf16_t* src = bB + (long)(k0 + kr) * ldb + nBase + c;
                union { uint2 u; bf16_t e[4]; } ub;
                ub.u = *reinterpret_cast<const uint2*>(src);
                Bs[buf][c + 0][kr] = ub.e[0];
                Bs[buf][c + 1][kr] = ub.e[1];
                Bs[buf][c + 2][kr] = ub.e[2];
                Bs[buf][c + 3][kr] = ub.e[3];
            }
        }
    };

    stage(0, 0);
    const int nk = K >> 5;
    for (int kt = 0; kt < nk; ++kt) {
        const int cur = kt & 1;
        const bool more = (kt + 1 < nk);
        if (more) stage(cur ^ 1, (kt + 1) << 5);      // prefetch next tile
#if HAS_ASYNC_LDS
        if (more) asm volatile("s_wait_asynccnt %0" :: "i"(PERWAVE));
        else      asm volatile("s_wait_asynccnt %0" :: "i"(0));
#endif
        __syncthreads();

        // A fragment 16x32: lanes 0-15 rows M, K = {kb..kb+7, kb+16..kb+23}
        const int arow = rb + (lane & 15);
        const int kb = (lane < 16) ? 0 : 8;
        union { uint4 q[2]; v16bf v; } ua;
        ua.q[0] = *reinterpret_cast<const uint4*>(&As[cur][arow][kb]);
        ua.q[1] = *reinterpret_cast<const uint4*>(&As[cur][arow][kb + 16]);

        const int kb2 = (lane < 16) ? 0 : 8;
        #pragma unroll
        for (int ct = 0; ct < CT; ++ct) {
            const int ncol = ct * 16 + (lane & 15);
            union { uint4 q[2]; v16bf v; } ubf;
            ubf.q[0] = *reinterpret_cast<const uint4*>(&Bs[cur][ncol][kb2]);
            ubf.q[1] = *reinterpret_cast<const uint4*>(&Bs[cur][ncol][kb2 + 16]);
            acc[ct] = __builtin_amdgcn_wmma_f32_16x16x32_bf16(
                false, ua.v, false, ubf.v, (short)0, acc[ct], false, false);
        }
        __syncthreads();
    }

    // C/D layout: VGPR r, lanes 0-15 -> M=r, lanes 16-31 -> M=8+r; N=lane&15
    const int rHi = (lane < 16) ? 0 : 8;
    #pragma unroll
    for (int ct = 0; ct < CT; ++ct) {
        const int gc = nBase + ct * 16 + (lane & 15);
        #pragma unroll
        for (int r = 0; r < 8; ++r) {
            const int gr = mBase + rb + rHi + r;
            const float v = acc[ct][r];
            const long oidx = (long)head * outZStep + (long)gr * ldout + gc;
            if constexpr (EPI == 0)      outB[oidx] = (bf16_t)(v + bias[gc]);
            else if constexpr (EPI == 1) outB[oidx] = (bf16_t)gelu_f(v + bias[gc]);
            else if constexpr (EPI == 2) outF[oidx] = v + bias[gc] + bias2[gc];
            else if constexpr (EPI == 3) outF[oidx] += v + bias[gc];
            else if constexpr (EPI == 4) outB[oidx] = (bf16_t)v;
            else if constexpr (EPI == 5) atomicAdd(&outF[oidx], v);
        }
    }
}

// ---------------------------------------------------------------------------
// small kernels
// ---------------------------------------------------------------------------
// f32 [mats][K][Nout] -> bf16 transposed [mats][Nout][K]
__global__ void cvtT_k(const float* __restrict__ in, bf16_t* __restrict__ out,
                       int K, int Nout, long total) {
    long i = (long)blockIdx.x * 256 + threadIdx.x;
    if (i >= total) return;
    long ms  = (long)K * Nout;
    long mat = i / ms, rem = i - mat * ms;
    int k = (int)(rem / Nout), n = (int)(rem - (long)k * Nout);
    out[mat * ms + (long)n * K + k] = (bf16_t)in[i];
}

__global__ void zero_k(float* __restrict__ p, int n) {
    int i = blockIdx.x * 256 + threadIdx.x;
    if (i < n) p[i] = 0.f;
}

// preprocess layer 1: gelu(x @ pre_w1 + b1), K=3 -> VALU
__global__ void pre1_k(const float* __restrict__ x, const float* __restrict__ w1,
                       const float* __restrict__ b1, bf16_t* __restrict__ out) {
    long i = (long)blockIdx.x * 256 + threadIdx.x;     // [0, N*512)
    int n = (int)(i >> 9), j = (int)(i & 511);
    float v = b1[j] + x[n * 3 + 0] * w1[j]
                    + x[n * 3 + 1] * w1[512 + j]
                    + x[n * 3 + 2] * w1[1024 + j];
    out[i] = (bf16_t)gelu_f(v);
}

// LayerNorm over D=256, wave per row, 8 rows / block
__global__ void ln_k(const float* __restrict__ xin, const float* __restrict__ g,
                     const float* __restrict__ b, bf16_t* __restrict__ out, int nrows) {
    int lane = threadIdx.x & 31, wid = threadIdx.x >> 5;
    int row = blockIdx.x * 8 + wid;
    if (row >= nrows) return;
    const float* xr = xin + (long)row * 256;
    float vals[8], s = 0.f, s2 = 0.f;
    #pragma unroll
    for (int i = 0; i < 8; ++i) {
        float v = xr[lane + i * 32];
        vals[i] = v; s += v; s2 += v * v;
    }
    #pragma unroll
    for (int o = 16; o; o >>= 1) { s += __shfl_xor(s, o, 32); s2 += __shfl_xor(s2, o, 32); }
    float m = s * (1.f / 256.f);
    float rs = rsqrtf(s2 * (1.f / 256.f) - m * m + 1e-5f);
    #pragma unroll
    for (int i = 0; i < 8; ++i) {
        int c = lane + i * 32;
        out[(long)row * 256 + c] = (bf16_t)((vals[i] - m) * rs * g[c] + b[c]);
    }
}

// softmax over 128 features (in place on bf16 logits), wave per (h,n) row
__global__ void softmax_q_k(bf16_t* __restrict__ q, const float* __restrict__ tq, int npts) {
    int lane = threadIdx.x & 31, wid = threadIdx.x >> 5;
    long row = (long)blockIdx.x * 8 + wid;             // [0, H*N)
    int h = (int)(row / npts);
    float t = clampt(tq[h]);
    bf16_t* qr = q + row * 128;
    float v[4], mx = -1e30f;
    #pragma unroll
    for (int i = 0; i < 4; ++i) { v[i] = (float)qr[lane + i * 32] / t; mx = fmaxf(mx, v[i]); }
    #pragma unroll
    for (int o = 16; o; o >>= 1) mx = fmaxf(mx, __shfl_xor(mx, o, 32));
    float s = 0.f;
    #pragma unroll
    for (int i = 0; i < 4; ++i) { v[i] = __expf(v[i] - mx); s += v[i]; }
    #pragma unroll
    for (int o = 16; o; o >>= 1) s += __shfl_xor(s, o, 32);
    float inv = 1.f / s;
    #pragma unroll
    for (int i = 0; i < 4; ++i) qr[lane + i * 32] = (bf16_t)(v[i] * inv);
}

// e = exp(k/tk) written transposed et[h][d][n]; per-feature denominators via
// LDS partials + one global atomicAdd per (block,d).
__global__ void ek_k(const bf16_t* __restrict__ kl, bf16_t* __restrict__ et,
                     float* __restrict__ kden, const float* __restrict__ tk, int npts) {
    __shared__ float sden[128];
    int t = threadIdx.x, h = blockIdx.y;
    int n = blockIdx.x * 256 + t;
    if (t < 128) sden[t] = 0.f;
    __syncthreads();
    float tkv = clampt(tk[h]);
    const bf16_t* kr = kl + ((long)h * npts + n) * 128;
    bf16_t* eb = et + (long)h * 128 * npts + n;
    for (int d = 0; d < 128; ++d) {
        float e = __expf((float)kr[d] / tkv);
        eb[(long)d * npts] = (bf16_t)e;                // coalesced across n
        atomicAdd(&sden[d], e);
    }
    __syncthreads();
    if (t < 128) atomicAdd(&kden[h * 128 + t], sden[t]);
}

// kvnT[h][c][d] = kv[h][d][c] / kden[h][d]  (transposed for BT=1 GEMM)
__global__ void kvn_k(const float* __restrict__ kv, const float* __restrict__ kden,
                      bf16_t* __restrict__ kvnT) {
    int i = blockIdx.x * 256 + threadIdx.x;            // i = h*4096 + d*32 + c
    if (i >= 8 * 128 * 32) return;
    int h = i >> 12, d = (i >> 5) & 127, c = i & 31;
    kvnT[(h << 12) + (c << 7) + d] = (bf16_t)(kv[i] / kden[(h << 7) + d]);
}

// head: out[n] = hb[n,:] . head_w + head_b, wave per row
__global__ void head_k(const bf16_t* __restrict__ hb, const float* __restrict__ w,
                       const float* __restrict__ b, float* __restrict__ out, int nrows) {
    int lane = threadIdx.x & 31, wid = threadIdx.x >> 5;
    int row = blockIdx.x * 8 + wid;
    if (row >= nrows) return;
    const bf16_t* hr = hb + (long)row * 256;
    float s = 0.f;
    #pragma unroll
    for (int i = 0; i < 8; ++i) { int c = lane + i * 32; s += (float)hr[c] * w[c]; }
    #pragma unroll
    for (int o = 16; o; o >>= 1) s += __shfl_xor(s, o, 32);
    if (lane == 0) out[row] = s + b[0];
}

// ---------------------------------------------------------------------------
extern "C" void kernel_launch(void* const* d_in, const int* in_sizes, int n_in,
                              void* d_out, int out_size, void* d_ws, size_t ws_size,
                              hipStream_t stream) {
    constexpr int N = 32768, D = 256, H = 8, DH = 32, QKD = 128, L = 5;

    const float* x           = (const float*)d_in[0];
    const float* pre_w1      = (const float*)d_in[1];
    const float* pre_b1      = (const float*)d_in[2];
    const float* pre_w2      = (const float*)d_in[3];
    const float* pre_b2      = (const float*)d_in[4];
    const float* placeholder = (const float*)d_in[5];
    const float* ln1_g       = (const float*)d_in[6];
    const float* ln1_b       = (const float*)d_in[7];
    const float* inp_w       = (const float*)d_in[8];
    const float* inp_b       = (const float*)d_in[9];
    const float* temp_q      = (const float*)d_in[10];
    const float* temp_k      = (const float*)d_in[11];
    const float* wq          = (const float*)d_in[12];
    const float* wk          = (const float*)d_in[13];
    const float* wv          = (const float*)d_in[14];
    const float* out_w1      = (const float*)d_in[15];
    const float* out_b1      = (const float*)d_in[16];
    const float* out_w2      = (const float*)d_in[17];
    const float* out_b2      = (const float*)d_in[18];
    const float* ln2_g       = (const float*)d_in[19];
    const float* ln2_b       = (const float*)d_in[20];
    const float* mlp_w1      = (const float*)d_in[21];
    const float* mlp_b1      = (const float*)d_in[22];
    const float* mlp_w2      = (const float*)d_in[23];
    const float* mlp_b2      = (const float*)d_in[24];
    const float* ln3_g       = (const float*)d_in[25];
    const float* ln3_b       = (const float*)d_in[26];
    const float* head_w      = (const float*)d_in[27];
    const float* head_b      = (const float*)d_in[28];

    // -------- workspace carve-out (256B aligned regions) --------
    char* base = (char*)d_ws;
    size_t off = 0;
    auto carve = [&](size_t bytes) -> char* {
        char* p = base + off;
        off += (bytes + 255) & ~(size_t)255;
        return p;
    };
    float*  fx   = (float*)  carve((size_t)N * D * 4);           // residual
    bf16_t* st1  = (bf16_t*) carve((size_t)N * 512 * 2);         // staging A
    bf16_t* st2  = (bf16_t*) carve((size_t)N * D * 2);           // staging B
    bf16_t* qb   = (bf16_t*) carve((size_t)H * N * QKD * 2);     // q
    bf16_t* kb   = (bf16_t*) carve((size_t)H * N * QKD * 2);     // k logits
    bf16_t* et   = (bf16_t*) carve((size_t)H * QKD * N * 2);     // exp(k/t)^T
    bf16_t* vb   = (bf16_t*) carve((size_t)H * N * DH * 2);      // v
    float*  kv   = (float*)  carve((size_t)(H * QKD * DH + H * QKD) * 4);
    float*  kden = kv + H * QKD * DH;
    bf16_t* kvnT = (bf16_t*) carve((size_t)H * QKD * DH * 2);
    bf16_t* pw2T = (bf16_t*) carve((size_t)512 * 256 * 2);       // [256][512]
    bf16_t* inpT = (bf16_t*) carve((size_t)L * 65536 * 2);       // [256][256]
    bf16_t* wqT  = (bf16_t*) carve((size_t)L * 4096 * 2);        // [128][32]
    bf16_t* wkT  = (bf16_t*) carve((size_t)L * 4096 * 2);
    bf16_t* wvT  = (bf16_t*) carve((size_t)L * 1024 * 2);        // [32][32]
    bf16_t* ow1T = (bf16_t*) carve((size_t)L * 65536 * 2);
    bf16_t* ow2T = (bf16_t*) carve((size_t)L * 65536 * 2);
    bf16_t* mw1T = (bf16_t*) carve((size_t)L * 65536 * 2);
    bf16_t* mw2T = (bf16_t*) carve((size_t)L * 65536 * 2);

    auto cvtT = [&](const float* src, bf16_t* dst, int K, int Nout, long total) {
        cvtT_k<<<(unsigned)((total + 255) / 256), 256, 0, stream>>>(src, dst, K, Nout, total);
    };
    cvtT(pre_w2, pw2T, 512, 256, 512 * 256);
    cvtT(inp_w,  inpT, 256, 256, (long)L * 65536);
    cvtT(wq,     wqT,   32, 128, (long)L * 4096);
    cvtT(wk,     wkT,   32, 128, (long)L * 4096);
    cvtT(wv,     wvT,   32,  32, (long)L * 1024);
    cvtT(out_w1, ow1T, 256, 256, (long)L * 65536);
    cvtT(out_w2, ow2T, 256, 256, (long)L * 65536);
    cvtT(mlp_w1, mw1T, 256, 256, (long)L * 65536);
    cvtT(mlp_w2, mw2T, 256, 256, (long)L * 65536);

    // -------- preprocess --------
    pre1_k<<<(N * 512) / 256, 256, 0, stream>>>(x, pre_w1, pre_b1, st1);
    gemm_k<128, 64, 2, 1><<<dim3(N / 128, 4, 1), 256, 0, stream>>>(
        st1, 0, 512, pw2T, 0, 512, 512, 1, fx, nullptr, 0, 256, pre_b2, placeholder);

    // -------- layers --------
    for (int i = 0; i < L; ++i) {
        ln_k<<<N / 8, 256, 0, stream>>>(fx, ln1_g + i * D, ln1_b + i * D, st1, N);
        // xm = LN(fx) @ inp_w + b -> st2 (bf16)
        gemm_k<128, 64, 0, 1><<<dim3(N / 128, 4, 1), 256, 0, stream>>>(
            st1, 0, 256, inpT + (size_t)i * 65536, 0, 256, 256, 1,
            nullptr, st2, 0, 256, inp_b + i * D, nullptr);
        // per-head q/k logits (K=32 -> 128) and v (K=32 -> 32); grid.z = heads
        gemm_k<128, 64, 4, 1><<<dim3(N / 128, 2, H), 256, 0, stream>>>(
            st2, 32, 256, wqT + (size_t)i * 4096, 0, 32, 32, 1,
            nullptr, qb, (long)N * 128, 128, nullptr, nullptr);
        gemm_k<128, 64, 4, 1><<<dim3(N / 128, 2, H), 256, 0, stream>>>(
            st2, 32, 256, wkT + (size_t)i * 4096, 0, 32, 32, 1,
            nullptr, kb, (long)N * 128, 128, nullptr, nullptr);
        gemm_k<128, 32, 4, 1><<<dim3(N / 128, 1, H), 256, 0, stream>>>(
            st2, 32, 256, wvT + (size_t)i * 1024, 0, 32, 32, 1,
            nullptr, vb, (long)N * 32, 32, nullptr, nullptr);
        softmax_q_k<<<(H * N) / 8, 256, 0, stream>>>(qb, temp_q + i * H, N);
        zero_k<<<(H * QKD * DH + H * QKD + 255) / 256, 256, 0, stream>>>(
            kv, H * QKD * DH + H * QKD);
        ek_k<<<dim3(N / 256, H), 256, 0, stream>>>(kb, et, kden, temp_k + i * H, N);
        // kv[h] = et[h](128xN) @ v[h](Nx32), split-K (8x4096) + f32 atomics
        gemm_k<128, 32, 5, 0><<<dim3(1, 1, H * 8), 256, 0, stream>>>(
            et, (long)QKD * N, N, vb, (long)N * 32, 32, N / 8, 8,
            kv, nullptr, (long)QKD * 32, 32, nullptr, nullptr);
        kvn_k<<<(H * QKD * DH) / 256, 256, 0, stream>>>(kv, kden, kvnT);
        // qkv[h] = q[h](Nx128) @ kvn[h](128x32) -> st2[n, h*32 + c]
        gemm_k<128, 32, 4, 1><<<dim3(N / 128, 1, H), 256, 0, stream>>>(
            qb, (long)N * 128, 128, kvnT, (long)QKD * 32, 128, 128, 1,
            nullptr, st2, 32, 256, nullptr, nullptr);
        // attn MLP: st1 = gelu(st2 @ out_w1 + b1); fx += st1 @ out_w2 + b2
        gemm_k<128, 64, 1, 1><<<dim3(N / 128, 4, 1), 256, 0, stream>>>(
            st2, 0, 256, ow1T + (size_t)i * 65536, 0, 256, 256, 1,
            nullptr, st1, 0, 256, out_b1 + i * D, nullptr);
        gemm_k<128, 64, 3, 1><<<dim3(N / 128, 4, 1), 256, 0, stream>>>(
            st1, 0, 256, ow2T + (size_t)i * 65536, 0, 256, 256, 1,
            fx, nullptr, 0, 256, out_b2 + i * D, nullptr);
        // MLP block
        ln_k<<<N / 8, 256, 0, stream>>>(fx, ln2_g + i * D, ln2_b + i * D, st1, N);
        gemm_k<128, 64, 1, 1><<<dim3(N / 128, 4, 1), 256, 0, stream>>>(
            st1, 0, 256, mw1T + (size_t)i * 65536, 0, 256, 256, 1,
            nullptr, st2, 0, 256, mlp_b1 + i * D, nullptr);
        gemm_k<128, 64, 3, 1><<<dim3(N / 128, 4, 1), 256, 0, stream>>>(
            st2, 0, 256, mw2T + (size_t)i * 65536, 0, 256, 256, 1,
            fx, nullptr, 0, 256, mlp_b2 + i * D, nullptr);
    }

    // -------- head --------
    ln_k<<<N / 8, 256, 0, stream>>>(fx, ln3_g, ln3_b, st1, N);
    head_k<<<N / 8, 256, 0, stream>>>(st1, head_w, head_b, (float*)d_out, N);

    (void)in_sizes; (void)n_in; (void)out_size; (void)ws_size;
}
